// AggFeatureModel_75952201662966
// MI455X (gfx1250) — compile-verified, matrix operand order checked by reading, not run on hardware
//
#include <hip/hip_runtime.h>
#include <hip/hip_bf16.h>
#include <math.h>

typedef __attribute__((ext_vector_type(16))) _Float16 v16h;
typedef __attribute__((ext_vector_type(8)))  float    v8f;

#define EPSF   1e-9f
#define VA     200
#define VB     100
#define NCOLS  1809
#define LOGC1  1.71828182845904523536f   /* expm1(1.0) */

__device__ __forceinline__ float logify(float v) {
    float m = expm1f(fabsf(v));
    return (v > 0.f) ? m : ((v < 0.f) ? -m : 0.f);
}
__device__ __forceinline__ float safe_sqrt(float x) {
    return (x > 0.f) ? sqrtf(x) : 0.f;
}

// ---------------------------------------------------------------------------
// Kernel 1: per-row category aggregation via LDS atomics (memory-bound core).
// One block per row; 256 threads (8 wave32). b128 vectorized streaming loads.
// Writes all columns except 1,2,3.
// ---------------------------------------------------------------------------
__global__ void __launch_bounds__(256)
row_cat_agg(const float* __restrict__ amount,
            const int*   __restrict__ cat_a,
            const int*   __restrict__ cat_b,
            const int*   __restrict__ seq_lens,
            float*       __restrict__ out,
            int T)
{
    __shared__ unsigned cntA[VA];
    __shared__ float    sumA[VA], ssqA[VA];
    __shared__ unsigned cntB[VB];
    __shared__ float    sumB[VB], ssqB[VB];
    __shared__ unsigned distA, distB;

    const int tid = threadIdx.x;
    const int row = blockIdx.x;

    for (int v = tid; v < VA; v += 256) { cntA[v] = 0u; sumA[v] = 0.f; ssqA[v] = 0.f; }
    for (int v = tid; v < VB; v += 256) { cntB[v] = 0u; sumB[v] = 0.f; ssqB[v] = 0.f; }
    if (tid == 0) { distA = 0u; distB = 0u; }
    __syncthreads();

    const size_t base = (size_t)row * (size_t)T;
    const float4* amt4 = reinterpret_cast<const float4*>(amount + base);
    const int4*   ca4  = reinterpret_cast<const int4*>(cat_a + base);
    const int4*   cb4  = reinterpret_cast<const int4*>(cat_b + base);

    // 4 elements per thread per iteration; 256 threads -> 1024 elems/iter.
    for (int q = tid; q < (T >> 2); q += 256) {
        const float4 a = amt4[q];
        const int4   ca = ca4[q];
        const int4   cb = cb4[q];
        float v0 = logify(a.x), v1 = logify(a.y);
        float v2 = logify(a.z), v3 = logify(a.w);

        atomicAdd(&cntA[ca.x], 1u); atomicAdd(&sumA[ca.x], v0); atomicAdd(&ssqA[ca.x], v0 * v0);
        atomicAdd(&cntA[ca.y], 1u); atomicAdd(&sumA[ca.y], v1); atomicAdd(&ssqA[ca.y], v1 * v1);
        atomicAdd(&cntA[ca.z], 1u); atomicAdd(&sumA[ca.z], v2); atomicAdd(&ssqA[ca.z], v2 * v2);
        atomicAdd(&cntA[ca.w], 1u); atomicAdd(&sumA[ca.w], v3); atomicAdd(&ssqA[ca.w], v3 * v3);

        atomicAdd(&cntB[cb.x], 1u); atomicAdd(&sumB[cb.x], v0); atomicAdd(&ssqB[cb.x], v0 * v0);
        atomicAdd(&cntB[cb.y], 1u); atomicAdd(&sumB[cb.y], v1); atomicAdd(&ssqB[cb.y], v1 * v1);
        atomicAdd(&cntB[cb.z], 1u); atomicAdd(&sumB[cb.z], v2); atomicAdd(&ssqB[cb.z], v2 * v2);
        atomicAdd(&cntB[cb.w], 1u); atomicAdd(&sumB[cb.w], v3); atomicAdd(&ssqB[cb.w], v3 * v3);
    }
    __syncthreads();

    const float sl = (float)seq_lens[row];
    float* o = out + (size_t)row * NCOLS;

    unsigned myDA = 0, myDB = 0;
    // cat_a derived columns: val1 -> 4/204/404, val2 -> 907/1107/1307
    for (int v = tid; v < VA; v += 256) {
        const float raw = (float)cntA[v];
        const float cnt = (v == 0) ? 0.f : raw;
        if (v > 0 && raw > 0.f) myDA++;
        const float den  = cnt + EPSF;
        const float denN = fmaxf(cnt - 1.f, 0.f) + EPSF;
        const float es = sumA[v], eq = ssqA[v];
        o[4   + v] = cnt;
        o[204 + v] = es / den;
        float a1 = fmaxf(eq - es * es / den, 0.f);
        o[404 + v] = safe_sqrt(a1 / denN);
        const float es2 = LOGC1 * raw;
        const float eq2 = LOGC1 * LOGC1 * raw;
        o[907  + v] = cnt;
        o[1107 + v] = es2 / den;
        float a2 = fmaxf(eq2 - es2 * es2 / den, 0.f);
        o[1307 + v] = safe_sqrt(a2 / denN);
    }
    // cat_b derived columns: val1 -> 604/704/804, val2 -> 1507/1607/1707
    for (int v = tid; v < VB; v += 256) {
        const float raw = (float)cntB[v];
        const float cnt = (v == 0) ? 0.f : raw;
        if (v > 0 && raw > 0.f) myDB++;
        const float den  = cnt + EPSF;
        const float denN = fmaxf(cnt - 1.f, 0.f) + EPSF;
        const float es = sumB[v], eq = ssqB[v];
        o[604 + v] = cnt;
        o[704 + v] = es / den;
        float a1 = fmaxf(eq - es * es / den, 0.f);
        o[804 + v] = safe_sqrt(a1 / denN);
        const float es2 = LOGC1 * raw;
        const float eq2 = LOGC1 * LOGC1 * raw;
        o[1507 + v] = cnt;
        o[1607 + v] = es2 / den;
        float a2 = fmaxf(eq2 - es2 * es2 / den, 0.f);
        o[1707 + v] = safe_sqrt(a2 / denN);
    }
    if (myDA) atomicAdd(&distA, myDA);
    if (myDB) atomicAdd(&distB, myDB);
    __syncthreads();

    if (tid == 0) {
        o[0] = sl;
        const float s2 = LOGC1 * (float)T;                 // sum of constant val2
        o[904] = s2;
        o[905] = s2 / (sl + EPSF);
        float a = fmaxf(LOGC1 * LOGC1 * (float)T - s2 * s2 / (sl + EPSF), 0.f);
        o[906] = safe_sqrt(a / (fmaxf(sl - 1.f, 0.f) + EPSF));
        o[1807] = (float)distA;
        o[1808] = (float)distB;
    }
}

// ---------------------------------------------------------------------------
// Kernel 2: whole-row sum / sumsq of logify(amount) via WMMA (A[16x32] * ones).
// 16 rows per block; 8 waves split T; f16 inputs, f32 accumulate.
// Wave index is readfirstlane'd so the K-loop is scalar-controlled and EXEC
// is provably all-ones around the WMMAs. Writes columns 1 (sum), 2, 3.
// ---------------------------------------------------------------------------
__global__ void __launch_bounds__(256)
row_stats_wmma(const float* __restrict__ amount,
               const int*   __restrict__ seq_lens,
               float*       __restrict__ out,
               int T)
{
    __shared__ float redS[16];
    __shared__ float redQ[16];

    const int tid  = threadIdx.x;
    // Wave-uniform by construction; readfirstlane makes it SALU-visible so the
    // K-loop compiles to scalar control flow (EXEC untouched around WMMA).
    const int wave = __builtin_amdgcn_readfirstlane(tid >> 5);
    const int lane = tid & 31;
    const int m0   = blockIdx.x * 16;
    const int r    = lane & 15;          // A-matrix row handled by this lane
    const int koff = (lane >> 4) * 8;    // lanes 16-31 hold K 8..15 / 24..31

    const float* rp = amount + (size_t)(m0 + r) * (size_t)T;

    if (tid < 16) { redS[tid] = 0.f; redQ[tid] = 0.f; }
    __syncthreads();

    v16h ones;
#pragma unroll
    for (int i = 0; i < 16; ++i) ones[i] = (_Float16)1.0f;

    v8f cs = {};   // row sums accumulator (D/C matrix, f32)
    v8f cq = {};   // row sum-of-squares accumulator

    const int tslice = __builtin_amdgcn_readfirstlane(T >> 3);  // 8 waves split T
    const int tbeg   = wave * tslice;
    const int tend   = tbeg + tslice;
    for (int t0 = tbeg; t0 < tend; t0 += 32) {
        const float4* p0 = reinterpret_cast<const float4*>(rp + t0 + koff);
        const float4* p1 = reinterpret_cast<const float4*>(rp + t0 + 16 + koff);
        const float4 a0 = p0[0], a1 = p0[1];
        const float4 b0 = p1[0], b1 = p1[1];
        float x[16];
        x[0]  = a0.x; x[1]  = a0.y; x[2]  = a0.z; x[3]  = a0.w;
        x[4]  = a1.x; x[5]  = a1.y; x[6]  = a1.z; x[7]  = a1.w;
        x[8]  = b0.x; x[9]  = b0.y; x[10] = b0.z; x[11] = b0.w;
        x[12] = b1.x; x[13] = b1.y; x[14] = b1.z; x[15] = b1.w;

        v16h av, aq;
#pragma unroll
        for (int i = 0; i < 16; ++i) {
            const float lv = logify(x[i]);
            av[i] = (_Float16)lv;
            aq[i] = (_Float16)(lv * lv);
        }
        // D = A x ones + C  ->  every D column holds the per-row K-chunk sum
        cs = __builtin_amdgcn_wmma_f32_16x16x32_f16(
                 false, av, false, ones, (short)0, cs, false, false);
        cq = __builtin_amdgcn_wmma_f32_16x16x32_f16(
                 false, aq, false, ones, (short)0, cq, false, false);
    }

    // N==0 column lives in lane 0 (M = j) and lane 16 (M = 8 + j), VGPR j.
    if (lane == 0 || lane == 16) {
        const int mb = (lane == 0) ? 0 : 8;
#pragma unroll
        for (int j = 0; j < 8; ++j) {
            atomicAdd(&redS[mb + j], cs[j]);
            atomicAdd(&redQ[mb + j], cq[j]);
        }
    }
    __syncthreads();

    if (tid < 16) {
        const int row = m0 + tid;
        const float S  = redS[tid];
        const float Q  = redQ[tid];
        const float sl = (float)seq_lens[row];
        float* o = out + (size_t)row * NCOLS;
        o[1] = S;
        o[2] = S / (sl + EPSF);
        float a = fmaxf(Q - S * S / (sl + EPSF), 0.f);
        o[3] = safe_sqrt(a / (fmaxf(sl - 1.f, 0.f) + EPSF));
    }
}

// ---------------------------------------------------------------------------
extern "C" void kernel_launch(void* const* d_in, const int* in_sizes, int n_in,
                              void* d_out, int out_size, void* d_ws, size_t ws_size,
                              hipStream_t stream)
{
    (void)n_in; (void)out_size; (void)d_ws; (void)ws_size;

    const float* amount   = (const float*)d_in[0];
    const int*   cat_a    = (const int*)d_in[1];
    const int*   cat_b    = (const int*)d_in[2];
    const int*   seq_lens = (const int*)d_in[3];
    float*       out      = (float*)d_out;

    const int B = in_sizes[3];
    const int T = in_sizes[0] / B;

    // Workhorse: per-row LDS-atomic aggregation (all columns except 1..3).
    row_cat_agg<<<dim3(B), dim3(256), 0, stream>>>(amount, cat_a, cat_b,
                                                   seq_lens, out, T);
    // WMMA row reductions for columns 1..3 (16 rows per block).
    row_stats_wmma<<<dim3(B / 16), dim3(256), 0, stream>>>(amount, seq_lens,
                                                           out, T);
}
